// PM_DMNet_41257455845608
// MI455X (gfx1250) — matrix operand order repaired and checked by reading, not compiled, not run on hardware
//
#include <hip/hip_runtime.h>
#include <hip/hip_bf16.h>
#include <math.h>

// ---------------------------------------------------------------------------
// PM-DMNet forward for MI455X (gfx1250, wave32, WMMA).
// Dominant GEMMs (A@xs graph propagation, and the nv-weighted pool contraction
// rewritten as LHS(307x1300)@W(1300xO)) run on v_wmma_f32_16x16x32_f16 with
// fp32 accumulation. The 16-column B panel of each block is staged once into
// LDS (transposed) so the K loop reads B via ds_load_b128 instead of strided
// global u16 gathers. Epilogues (softmax adjacency, gates, tanh, attention
// softmax) are fp32 VALU. Decoder (batch 192) is chunked 12x16.
// ---------------------------------------------------------------------------

typedef _Float16 h16v __attribute__((ext_vector_type(16)));
typedef _Float16 h8v  __attribute__((ext_vector_type(8)));
typedef float    f8v  __attribute__((ext_vector_type(8)));

#define BB 16      // batch
#define TT 12      // encoder steps
#define QQ 12      // decoder horizon
#define NN 307     // nodes
#define NP 320     // nodes padded to mult of 16 (and 32 for WMMA K)
#define HH 64      // hidden
#define DD 10      // node-vector dim
#define TDIM 16    // time-embedding dim
#define CI 65      // cheb input channels (1 + H)
#define CP 80      // CI padded to mult of 16
#define KREAL 1300 // D*K*CI
#define KPAD 1312  // KREAL padded to mult of 32

__device__ __forceinline__ float reluf(float x) { return x > 0.f ? x : 0.f; }
__device__ __forceinline__ float sigf(float x) { return 1.f / (1.f + __expf(-x)); }

// ---------------------------------------------------------------------------
// Generic batched f16 GEMM on WMMA: C = A(MxK) @ B(KxNc). M=NP, K mult of 32,
// Nc mult of 16. Block = 128 threads = 4 waves, each wave owns one 16x16 tile
// in the same 16-column stripe; the Kdim x 16 B panel is staged to LDS
// transposed ([n][k]) so each lane's 32-wide B fragment is two 16B LDS loads.
// grid = (Nc/16, NP/64, batch); dynamic LDS = Kdim*16*2 bytes.
// STORE_HALF selects f16 or f32 output.
// ---------------------------------------------------------------------------
template <int STORE_HALF>
__global__ __launch_bounds__(128) void wmma_gemm_f16(
    const _Float16* __restrict__ A, long aBatch, int lda,
    const _Float16* __restrict__ Bm, long bBatch, int ldb,
    void* __restrict__ Cv, long cBatch, int ldc, int Kdim) {
  extern __shared__ _Float16 ldsT[];  // [16][Kdim] transposed B panel
  const int tid  = threadIdx.x;
  const int lane = tid & 31;
  const int wave = tid >> 5;
  const int l16  = lane & 15;
  const int hf   = lane >> 4;
  const int row0 = (blockIdx.y * 4 + wave) * 16;
  const int n0   = blockIdx.x * 16;
  const long z   = blockIdx.z;
  const _Float16* Ab = A + z * aBatch;
  const _Float16* Bb = Bm + z * bBatch;

  // Cooperative stage of the B panel into LDS, transposed.  128 threads cover
  // 64 K-rows x 16 cols per pass (one coalesced 16B global load per thread).
  {
    const int krow = tid >> 1;
    const int ncol = (tid & 1) * 8;
    for (int k = krow; k < Kdim; k += 64) {
      h8v v = *(const h8v*)(Bb + (long)k * ldb + n0 + ncol);
#pragma unroll
      for (int j = 0; j < 8; ++j) ldsT[(ncol + j) * Kdim + k] = v[j];
    }
  }
  __syncthreads();

  f8v acc = {0.f, 0.f, 0.f, 0.f, 0.f, 0.f, 0.f, 0.f};
  const _Float16* bl = ldsT + (long)l16 * Kdim + 8 * hf;
  for (int kb = 0; kb < Kdim; kb += 32) {
    // A fragment (16x32 f16): lane(m=l16,h=hf), K = 8h+e (e<8) / 8h+16+(e-8)
    const _Float16* ar = Ab + (long)(row0 + l16) * lda + kb + 8 * hf;
    h8v a0 = *(const h8v*)(ar);
    h8v a1 = *(const h8v*)(ar + 16);
    if (kb + 32 < Kdim) __builtin_prefetch(ar + 32, 0, 1);  // global_prefetch_b8
    h16v af;
#pragma unroll
    for (int j = 0; j < 8; ++j) { af[j] = a0[j]; af[8 + j] = a1[j]; }
    // B fragment (32x16 f16): lane(n=l16,h=hf), K = 8h+e (e<8) / 8h+16+(e-8)
    h8v b0 = *(const h8v*)(bl + kb);
    h8v b1 = *(const h8v*)(bl + kb + 16);
    h16v bf;
#pragma unroll
    for (int j = 0; j < 8; ++j) { bf[j] = b0[j]; bf[8 + j] = b1[j]; }
    acc = __builtin_amdgcn_wmma_f32_16x16x32_f16(false, af, false, bf,
                                                 (short)0, acc, false, false);
  }
  if (STORE_HALF) {
    _Float16* C = (_Float16*)Cv + z * cBatch;
#pragma unroll
    for (int j = 0; j < 8; ++j)
      C[(long)(row0 + j + 8 * hf) * ldc + n0 + l16] = (_Float16)acc[j];
  } else {
    float* C = (float*)Cv + z * cBatch;
#pragma unroll
    for (int j = 0; j < 8; ++j)
      C[(long)(row0 + j + 8 * hf) * ldc + n0 + l16] = acc[j];
  }
}

// ------------------------- small support kernels ---------------------------

__global__ void zero_f32_kernel(float* p, long n) {
  long i = (long)blockIdx.x * blockDim.x + threadIdx.x;
  if (i < n) p[i] = 0.f;
}

// ne[b,t,:] = Temb[int(tod*288)] * Demb[int(dow)]
__global__ void ne_kernel(const float* __restrict__ st, const float* __restrict__ Temb,
                          const float* __restrict__ Demb, float* __restrict__ ne, int steps) {
  int i = blockIdx.x * blockDim.x + threadIdx.x;
  if (i >= BB * steps * TDIM) return;
  int d  = i % TDIM;
  int bt = i / TDIM;
  const float* cell = st + (long)bt * NN * 3;  // [b,t,0,:]
  int it = (int)(cell[1] * 288.0f); it = it < 0 ? 0 : (it > 287 ? 287 : it);
  int id = (int)cell[2];            id = id < 0 ? 0 : (id > 6 ? 6 : id);
  ne[i] = Temb[it * TDIM + d] * Demb[id * TDIM + d];
}

// wpool (D,K,CI,O) contiguous -> f16 (KPAD,O), zero padded rows
__global__ void pack_w_kernel(const float* __restrict__ w, _Float16* __restrict__ wp, int O) {
  int i = blockIdx.x * blockDim.x + threadIdx.x;
  if (i >= KPAD * O) return;
  int r = i / O, o = i % O;
  float v = (r < KREAL) ? w[(long)r * O + o] : 0.f;
  wp[(long)r * O + o] = (_Float16)v;
}

// f32 (rows x cols, srcLd, from rowOff) -> f16 (rows x cols)
__global__ void cvt_rows_kernel(const float* __restrict__ src, _Float16* __restrict__ dst,
                                int rows, int cols, int rowOff, int srcLd) {
  int i = blockIdx.x * blockDim.x + threadIdx.x;
  if (i >= rows * cols) return;
  int r = i / cols, c = i % cols;
  dst[i] = (_Float16)src[(long)(r + rowOff) * srcLd + c];
}

// out[r,o] = emb[r,:16] @ w[0:16,:] + b   (STE half of attention projections)
__global__ void ste_proj_kernel(const float* __restrict__ emb, int rowStride,
                                const float* __restrict__ w, const float* __restrict__ b,
                                float* __restrict__ out, int rows) {
  int i = blockIdx.x * blockDim.x + threadIdx.x;
  if (i >= rows * HH) return;
  int r = i / HH, o = i % HH;
  float s = b[o];
#pragma unroll
  for (int k = 0; k < TDIM; ++k) s += emb[(long)r * rowStride + k] * w[k * HH + o];
  out[i] = s;
}

// nv[b,n,d] = tanh(E[n,d] * (emb[b] @ fcw + fcb)[d])
__global__ void nv_kernel(const float* __restrict__ emb, int embStride,
                          const float* __restrict__ fcw, const float* __restrict__ fcb,
                          const float* __restrict__ E, float* __restrict__ nv, int NB) {
  int i = blockIdx.x * blockDim.x + threadIdx.x;
  if (i >= NB * NN * DD) return;
  int d = i % DD, n = (i / DD) % NN, b = i / (DD * NN);
  float g = fcb[d];
#pragma unroll
  for (int k = 0; k < TDIM; ++k) g += emb[(long)b * embStride + k] * fcw[k * DD + d];
  nv[i] = tanhf(E[n * DD + d] * g);
}

// A[b,n,:] = softmax(relu(nv[b,n] . nv[b,m])) over m, stored f16 padded NPxNP
__global__ __launch_bounds__(NP) void adj_kernel(const float* __restrict__ nv,
                                                 _Float16* __restrict__ A, int NB) {
  int bn = blockIdx.x;
  int n  = bn % NP;
  int b  = bn / NP;
  int m  = threadIdx.x;
  __shared__ float sm[NP];
  float s = -1e30f;
  if (n < NN && m < NN) {
    const float* pa = nv + ((long)b * NN + n) * DD;
    const float* pc = nv + ((long)b * NN + m) * DD;
    float acc = 0.f;
#pragma unroll
    for (int d = 0; d < DD; ++d) acc += pa[d] * pc[d];
    s = reluf(acc);
  }
  sm[m] = s;
  __syncthreads();
  for (int st = 256; st > 0; st >>= 1) {
    if (m < st && m + st < NP) sm[m] = fmaxf(sm[m], sm[m + st]);
    __syncthreads();
  }
  float mx = sm[0];
  __syncthreads();
  float e = (n < NN && m < NN) ? __expf(s - mx) : 0.f;
  sm[m] = e;
  __syncthreads();
  for (int st = 256; st > 0; st >>= 1) {
    if (m < st && m + st < NP) sm[m] += sm[m + st];
    __syncthreads();
  }
  float inv = (n < NN) ? 1.f / sm[0] : 0.f;
  A[((long)b * NP + n) * NP + m] = (_Float16)(e * inv);
}

// xs f16 (NB,NP,CP): col0 = x (or 0), cols 1..64 = state (optionally *z), pad 0
__global__ void xs_pack_kernel(const float* __restrict__ x, long xB, int xN,
                               const float* __restrict__ state, const float* __restrict__ zg,
                               _Float16* __restrict__ xs, int NB) {
  long i = (long)blockIdx.x * blockDim.x + threadIdx.x;
  if (i >= (long)NB * NP * CP) return;
  int c = (int)(i % CP);
  int n = (int)((i / CP) % NP);
  int b = (int)(i / ((long)CP * NP));
  float v = 0.f;
  if (n < NN) {
    if (c == 0) {
      v = x ? x[(long)b * xB + (long)n * xN] : 0.f;
    } else if (c <= HH) {
      long si = ((long)b * NN + n) * HH + (c - 1);
      v = state[si];
      if (zg) v *= zg[si];
    }
  }
  xs[i] = (_Float16)v;
}

// LHS f16 (NB,NP,KPAD): LHS[n, d*130+k*65+i] = nv[n,d] * (k ? xg : xs)[n,i]
__global__ void lhs_kernel(const float* __restrict__ nv, const _Float16* __restrict__ xs,
                           const _Float16* __restrict__ xg, _Float16* __restrict__ lhs, int NB) {
  long i = (long)blockIdx.x * blockDim.x + threadIdx.x;
  if (i >= (long)NB * NP * KPAD) return;
  int r = (int)(i % KPAD);
  int n = (int)((i / KPAD) % NP);
  int b = (int)(i / ((long)KPAD * NP));
  float v = 0.f;
  if (n < NN && r < KREAL) {
    int d = r / 130, rem = r % 130;
    int k = rem / 65, ii = rem % 65;
    const _Float16* src = k ? xg : xs;
    v = nv[((long)b * NN + n) * DD + d] * (float)src[((long)b * NP + n) * CP + ii];
  }
  lhs[i] = (_Float16)v;
}

// z,r = sigmoid(Cg + nv @ bpool), split halves
__global__ void gate_kernel(const float* __restrict__ Cg, const float* __restrict__ nv,
                            const float* __restrict__ bpool, float* __restrict__ z,
                            float* __restrict__ r, int NB) {
  int i = blockIdx.x * blockDim.x + threadIdx.x;
  if (i >= NB * NN * HH) return;
  int o = i % HH, n = (i / HH) % NN, b = i / (HH * NN);
  const float* nvp = nv + ((long)b * NN + n) * DD;
  const float* cg  = Cg + ((long)b * NP + n) * 128;
  float az = cg[o], ar = cg[HH + o];
#pragma unroll
  for (int d = 0; d < DD; ++d) {
    az += nvp[d] * bpool[d * 128 + o];
    ar += nvp[d] * bpool[d * 128 + HH + o];
  }
  z[i] = sigf(az);
  r[i] = sigf(ar);
}

// snew = r*sold + (1-r)*tanh(Cu + nv @ bpool)
__global__ void upd_kernel(const float* __restrict__ Cu, const float* __restrict__ nv,
                           const float* __restrict__ bpool, const float* __restrict__ r,
                           const float* __restrict__ sold, float* __restrict__ snew, int NB) {
  int i = blockIdx.x * blockDim.x + threadIdx.x;
  if (i >= NB * NN * HH) return;
  int o = i % HH, n = (i / HH) % NN, b = i / (HH * NN);
  const float* nvp = nv + ((long)b * NN + n) * DD;
  float h = Cu[((long)b * NP + n) * HH + o];
#pragma unroll
  for (int d = 0; d < DD; ++d) h += nvp[d] * bpool[d * HH + o];
  h = tanhf(h);
  float rv = r[i];
  snew[i] = rv * sold[i] + (1.f - rv) * h;
}

// state f32 (BB,NN,HH) -> f16 (BB,NP,HH), padded rows zero
__global__ void h16_kernel(const float* __restrict__ s, _Float16* __restrict__ h16) {
  long i = (long)blockIdx.x * blockDim.x + threadIdx.x;
  if (i >= (long)BB * NP * HH) return;
  int o = (int)(i % HH);
  int n = (int)((i / HH) % NP);
  int b = (int)(i / ((long)HH * NP));
  float v = (n < NN) ? s[((long)b * NN + n) * HH + o] : 0.f;
  h16[i] = (_Float16)v;
}

// Fused attention: q/k/v relu, per-head q.k scores, softmax over q (axis=-2 of
// bhnqp with P=1), out16[b,q,n,:] = att[q,h] * v.  One block per (b,n), 64 thr.
__global__ __launch_bounds__(64) void attn_kernel(
    const float* __restrict__ Pq, const float* __restrict__ Pk, const float* __restrict__ Pv,
    const float* __restrict__ sq, const float* __restrict__ sk, const float* __restrict__ sv,
    _Float16* __restrict__ out16) {
  int bn = blockIdx.x;
  int n  = bn % NP;
  int b  = bn / NP;
  int o  = threadIdx.x;
  if (n >= NN) {
    for (int q = 0; q < QQ; ++q)
      out16[(((long)(b * QQ + q)) * NP + n) * HH + o] = (_Float16)0.f;
    return;
  }
  long base = ((long)b * NP + n) * HH + o;
  float kk = reluf(Pk[base] + sk[b * HH + o]);
  float vv = reluf(Pv[base] + sv[b * HH + o]);
  float pq = Pq[base];
  __shared__ float satt[QQ][8];
  for (int q = 0; q < QQ; ++q) {
    float qv = reluf(pq + sq[(long)(b * QQ + q) * HH + o]);
    float p  = qv * kk;
    p += __shfl_xor(p, 1, 32);
    p += __shfl_xor(p, 2, 32);
    p += __shfl_xor(p, 4, 32);
    if ((o & 7) == 0) satt[q][o >> 3] = p;
  }
  __syncthreads();
  if (o < 8) {  // softmax over q per head
    float mx = -1e30f;
    for (int q = 0; q < QQ; ++q) mx = fmaxf(mx, satt[q][o]);
    float s = 0.f;
    for (int q = 0; q < QQ; ++q) { float e = __expf(satt[q][o] - mx); satt[q][o] = e; s += e; }
    float inv = 1.f / s;
    for (int q = 0; q < QQ; ++q) satt[q][o] *= inv;
  }
  __syncthreads();
  int h = o >> 3;
  for (int q = 0; q < QQ; ++q)
    out16[(((long)(b * QQ + q)) * NP + n) * HH + o] = (_Float16)(satt[q][h] * vv);
}

// de[bq,n,o] = Xw[b,n,o] + OutF[bq,n,o] + b2[o]
__global__ void de_kernel(const float* __restrict__ Xw, const float* __restrict__ OutF,
                          const float* __restrict__ b2, float* __restrict__ de) {
  long i = (long)blockIdx.x * blockDim.x + threadIdx.x;
  if (i >= (long)BB * QQ * NN * HH) return;
  int o  = (int)(i % HH);
  int n  = (int)((i / HH) % NN);
  int bq = (int)(i / ((long)HH * NN));
  int b  = bq / QQ;
  de[i] = Xw[((long)b * NP + n) * HH + o] + OutF[((long)bq * NP + n) * HH + o] + b2[o];
}

// out[b,q,n] = dstate[bq,n,:] @ w_out[q,:,0] + b_out[q]
__global__ void out_kernel(const float* __restrict__ ds, const float* __restrict__ w_out,
                           const float* __restrict__ b_out, float* __restrict__ out) {
  int i = blockIdx.x * blockDim.x + threadIdx.x;
  if (i >= BB * QQ * NN) return;
  int n = i % NN, bq = i / NN, q = bq % QQ;
  const float* sp = ds + ((long)bq * NN + n) * HH;
  float s = b_out[q];
#pragma unroll
  for (int h = 0; h < HH; ++h) s += sp[h] * w_out[q * HH + h];
  out[i] = s;
}

// ---------------------------------------------------------------------------
// Host orchestration
// ---------------------------------------------------------------------------
static inline int cdiv(long a, long b) { return (int)((a + b - 1) / b); }

extern "C" void kernel_launch(void* const* d_in, const int* in_sizes, int n_in,
                              void* d_out, int out_size, void* d_ws, size_t ws_size,
                              hipStream_t stream) {
  // Input order: jax tree_flatten (sorted dict keys at every level) of
  // {'source','traget','params'}: params first ('params'<'source'<'traget'),
  // params keys sorted: Demb,E,Temb,b_out,dec{b_g,b_u,fc_g_b,fc_g_w,fc_u_b,
  // fc_u_w,w_g,w_u},enc{...},ta{b2,k_b,k_w,q_b,q_w,v_b,v_w,w2},w_out.
  const float* Demb   = (const float*)d_in[0];   // (7,16)
  const float* E      = (const float*)d_in[1];   // (307,10)
  const float* Temb   = (const float*)d_in[2];   // (288,16)
  const float* b_out  = (const float*)d_in[3];   // (12,1)
  const float* dbg    = (const float*)d_in[4];   // dec.b_g (10,128)
  const float* dbu    = (const float*)d_in[5];   // dec.b_u (10,64)
  const float* dfgb   = (const float*)d_in[6];   // dec.fc_g_b (10)
  const float* dfgw   = (const float*)d_in[7];   // dec.fc_g_w (16,10)
  const float* dfub   = (const float*)d_in[8];
  const float* dfuw   = (const float*)d_in[9];
  const float* dwg    = (const float*)d_in[10];  // dec.w_g (10,2,65,128)
  const float* dwu    = (const float*)d_in[11];  // dec.w_u (10,2,65,64)
  const float* ebg    = (const float*)d_in[12];
  const float* ebu    = (const float*)d_in[13];
  const float* efgb   = (const float*)d_in[14];
  const float* efgw   = (const float*)d_in[15];
  const float* efub   = (const float*)d_in[16];
  const float* efuw   = (const float*)d_in[17];
  const float* ewg    = (const float*)d_in[18];
  const float* ewu    = (const float*)d_in[19];
  const float* ta_b2  = (const float*)d_in[20];
  const float* ta_kb  = (const float*)d_in[21];
  const float* ta_kw  = (const float*)d_in[22];  // (80,64)
  const float* ta_qb  = (const float*)d_in[23];
  const float* ta_qw  = (const float*)d_in[24];
  const float* ta_vb  = (const float*)d_in[25];
  const float* ta_vw  = (const float*)d_in[26];
  const float* ta_w2  = (const float*)d_in[27];  // (2,64,64)
  const float* w_out  = (const float*)d_in[28];  // (12,64,1)
  const float* source = (const float*)d_in[29];  // (16,12,307,3)
  const float* traget = (const float*)d_in[30];  // (16,12,307,3)
  (void)in_sizes; (void)n_in; (void)out_size; (void)ws_size;

  // ---- workspace carve-out ----
  char* wsb = (char*)d_ws;
  size_t off = 0;
  auto alloc = [&](size_t bytes) -> void* {
    off = (off + 255) & ~(size_t)255;
    void* p = wsb + off;
    off += bytes;
    return p;
  };
  float*     ne1   = (float*)alloc(sizeof(float) * BB * TT * TDIM);
  float*     ne2   = (float*)alloc(sizeof(float) * BB * QQ * TDIM);
  _Float16*  WgE   = (_Float16*)alloc(2ul * KPAD * 128);
  _Float16*  WuE   = (_Float16*)alloc(2ul * KPAD * 64);
  _Float16*  WgD   = (_Float16*)alloc(2ul * KPAD * 128);
  _Float16*  WuD   = (_Float16*)alloc(2ul * KPAD * 64);
  _Float16*  qw16  = (_Float16*)alloc(2ul * 64 * 64);
  _Float16*  kw16  = (_Float16*)alloc(2ul * 64 * 64);
  _Float16*  vw16  = (_Float16*)alloc(2ul * 64 * 64);
  _Float16*  w20   = (_Float16*)alloc(2ul * 64 * 64);
  _Float16*  w21   = (_Float16*)alloc(2ul * 64 * 64);
  float*     sq    = (float*)alloc(sizeof(float) * BB * QQ * HH);
  float*     sk    = (float*)alloc(sizeof(float) * BB * HH);
  float*     sv    = (float*)alloc(sizeof(float) * BB * HH);
  float*     state = (float*)alloc(sizeof(float) * BB * NN * HH);
  float*     nv    = (float*)alloc(sizeof(float) * BB * NN * DD);
  _Float16*  A16   = (_Float16*)alloc(2ul * BB * NP * NP);
  _Float16*  xs16  = (_Float16*)alloc(2ul * BB * NP * CP);
  _Float16*  xg16  = (_Float16*)alloc(2ul * BB * NP * CP);
  _Float16*  lhs   = (_Float16*)alloc(2ul * BB * NP * KPAD);
  float*     Cg    = (float*)alloc(sizeof(float) * BB * NP * 128);
  float*     Cu    = (float*)alloc(sizeof(float) * BB * NP * 64);
  float*     zb    = (float*)alloc(sizeof(float) * BB * NN * HH);
  float*     rb    = (float*)alloc(sizeof(float) * BB * NN * HH);
  _Float16*  h16   = (_Float16*)alloc(2ul * BB * NP * HH);
  float*     Pq    = (float*)alloc(sizeof(float) * BB * NP * HH);
  float*     Pk    = (float*)alloc(sizeof(float) * BB * NP * HH);
  float*     Pv    = (float*)alloc(sizeof(float) * BB * NP * HH);
  _Float16*  out16 = (_Float16*)alloc(2ul * BB * QQ * NP * HH);
  float*     OutF  = (float*)alloc(sizeof(float) * BB * QQ * NP * HH);
  float*     Xw    = (float*)alloc(sizeof(float) * BB * NP * HH);
  float*     de    = (float*)alloc(sizeof(float) * BB * QQ * NN * HH);
  float*     dstate= (float*)alloc(sizeof(float) * BB * QQ * NN * HH);

  const int TB = 256;
  // ---- embeddings & weight packing ----
  ne_kernel<<<cdiv(BB * TT * TDIM, TB), TB, 0, stream>>>(source, Temb, Demb, ne1, TT);
  ne_kernel<<<cdiv(BB * QQ * TDIM, TB), TB, 0, stream>>>(traget, Temb, Demb, ne2, QQ);
  pack_w_kernel<<<cdiv((long)KPAD * 128, TB), TB, 0, stream>>>(ewg, WgE, 128);
  pack_w_kernel<<<cdiv((long)KPAD * 64, TB), TB, 0, stream>>>(ewu, WuE, 64);
  pack_w_kernel<<<cdiv((long)KPAD * 128, TB), TB, 0, stream>>>(dwg, WgD, 128);
  pack_w_kernel<<<cdiv((long)KPAD * 64, TB), TB, 0, stream>>>(dwu, WuD, 64);
  cvt_rows_kernel<<<cdiv(64 * 64, TB), TB, 0, stream>>>(ta_qw, qw16, 64, 64, 16, 64);
  cvt_rows_kernel<<<cdiv(64 * 64, TB), TB, 0, stream>>>(ta_kw, kw16, 64, 64, 16, 64);
  cvt_rows_kernel<<<cdiv(64 * 64, TB), TB, 0, stream>>>(ta_vw, vw16, 64, 64, 16, 64);
  cvt_rows_kernel<<<cdiv(64 * 64, TB), TB, 0, stream>>>(ta_w2, w20, 64, 64, 0, 64);
  cvt_rows_kernel<<<cdiv(64 * 64, TB), TB, 0, stream>>>(ta_w2 + 64 * 64, w21, 64, 64, 0, 64);
  ste_proj_kernel<<<cdiv(BB * QQ * HH, TB), TB, 0, stream>>>(ne2, TDIM, ta_qw, ta_qb, sq, BB * QQ);
  ste_proj_kernel<<<cdiv(BB * HH, TB), TB, 0, stream>>>(ne1 + (TT - 1) * TDIM, TT * TDIM, ta_kw, ta_kb, sk, BB);
  ste_proj_kernel<<<cdiv(BB * HH, TB), TB, 0, stream>>>(ne1 + (TT - 1) * TDIM, TT * TDIM, ta_vw, ta_vb, sv, BB);
  zero_f32_kernel<<<cdiv((long)BB * NN * HH, TB), TB, 0, stream>>>(state, (long)BB * NN * HH);

  // One _pm_cell pass (gate dgcn + update dgcn) at batch 16.
  auto pm_cell = [&](const float* emb, int embStride,
                     const float* fgw, const float* fgb, const float* fuw, const float* fub,
                     const _Float16* Wg, const _Float16* Wu,
                     const float* bg, const float* bu,
                     const float* x, long xB, int xN,
                     const float* sin, float* sout) {
    dim3 g5(CP / 16, NP / 64, BB), g8(128 / 16, NP / 64, BB), g4(64 / 16, NP / 64, BB);
    const size_t ldsNP = (size_t)NP * 16 * 2;    // A@xs B panel (K=NP)
    const size_t ldsKP = (size_t)KPAD * 16 * 2;  // LHS@W B panel (K=KPAD)
    // -- gate dgcn --
    nv_kernel<<<cdiv(BB * NN * DD, TB), TB, 0, stream>>>(emb, embStride, fgw, fgb, E, nv, BB);
    adj_kernel<<<BB * NP, NP, 0, stream>>>(nv, A16, BB);
    xs_pack_kernel<<<cdiv((long)BB * NP * CP, TB), TB, 0, stream>>>(x, xB, xN, sin, nullptr, xs16, BB);
    wmma_gemm_f16<1><<<g5, 128, ldsNP, stream>>>(A16, (long)NP * NP, NP, xs16, (long)NP * CP, CP,
                                                 xg16, (long)NP * CP, CP, NP);
    lhs_kernel<<<cdiv((long)BB * NP * KPAD, TB), TB, 0, stream>>>(nv, xs16, xg16, lhs, BB);
    wmma_gemm_f16<0><<<g8, 128, ldsKP, stream>>>(lhs, (long)NP * KPAD, KPAD, Wg, 0, 128,
                                                 Cg, (long)NP * 128, 128, KPAD);
    gate_kernel<<<cdiv(BB * NN * HH, TB), TB, 0, stream>>>(Cg, nv, bg, zb, rb, BB);
    // -- update dgcn --
    nv_kernel<<<cdiv(BB * NN * DD, TB), TB, 0, stream>>>(emb, embStride, fuw, fub, E, nv, BB);
    adj_kernel<<<BB * NP, NP, 0, stream>>>(nv, A16, BB);
    xs_pack_kernel<<<cdiv((long)BB * NP * CP, TB), TB, 0, stream>>>(x, xB, xN, sin, zb, xs16, BB);
    wmma_gemm_f16<1><<<g5, 128, ldsNP, stream>>>(A16, (long)NP * NP, NP, xs16, (long)NP * CP, CP,
                                                 xg16, (long)NP * CP, CP, NP);
    lhs_kernel<<<cdiv((long)BB * NP * KPAD, TB), TB, 0, stream>>>(nv, xs16, xg16, lhs, BB);
    wmma_gemm_f16<0><<<g4, 128, ldsKP, stream>>>(lhs, (long)NP * KPAD, KPAD, Wu, 0, 64,
                                                 Cu, (long)NP * 64, 64, KPAD);
    upd_kernel<<<cdiv(BB * NN * HH, TB), TB, 0, stream>>>(Cu, nv, bu, rb, sin, sout, BB);
  };

  // ---- encoder scan ----
  for (int t = 0; t < TT; ++t) {
    pm_cell(ne1 + t * TDIM, TT * TDIM, efgw, efgb, efuw, efub, WgE, WuE, ebg, ebu,
            source + (long)t * NN * 3, (long)TT * NN * 3, 3, state, state);
  }

  // ---- transform attention ----
  h16_kernel<<<cdiv((long)BB * NP * HH, TB), TB, 0, stream>>>(state, h16);
  {
    const size_t ldsHH = (size_t)HH * 16 * 2;  // K=64 B panel
    dim3 g4(64 / 16, NP / 64, BB);
    wmma_gemm_f16<0><<<g4, 128, ldsHH, stream>>>(h16, (long)NP * HH, HH, qw16, 0, 64,
                                                 Pq, (long)NP * HH, 64, 64);
    wmma_gemm_f16<0><<<g4, 128, ldsHH, stream>>>(h16, (long)NP * HH, HH, kw16, 0, 64,
                                                 Pk, (long)NP * HH, 64, 64);
    wmma_gemm_f16<0><<<g4, 128, ldsHH, stream>>>(h16, (long)NP * HH, HH, vw16, 0, 64,
                                                 Pv, (long)NP * HH, 64, 64);
  }
  attn_kernel<<<BB * NP, 64, 0, stream>>>(Pq, Pk, Pv, sq, sk, sv, out16);
  {
    const size_t ldsHH = (size_t)HH * 16 * 2;
    dim3 g4a(64 / 16, NP / 64, BB * QQ);
    wmma_gemm_f16<0><<<g4a, 128, ldsHH, stream>>>(out16, (long)NP * HH, HH, w21, 0, 64,
                                                  OutF, (long)NP * HH, 64, 64);
    dim3 g4b(64 / 16, NP / 64, BB);
    wmma_gemm_f16<0><<<g4b, 128, ldsHH, stream>>>(h16, (long)NP * HH, HH, w20, 0, 64,
                                                  Xw, (long)NP * HH, 64, 64);
  }
  de_kernel<<<cdiv((long)BB * QQ * NN * HH, TB), TB, 0, stream>>>(Xw, OutF, ta_b2, de);

  // ---- decoder cell (batch 192 chunked as 12 x 16) ----
  for (int c = 0; c < QQ; ++c) {
    long chunk = (long)c * BB * NN * HH;
    pm_cell(ne2 + (long)c * BB * TDIM, TDIM, dfgw, dfgb, dfuw, dfub, WgD, WuD, dbg, dbu,
            nullptr, 0, 0, de + chunk, dstate + chunk);
  }

  // ---- output projection ----
  out_kernel<<<cdiv(BB * QQ * NN, TB), TB, 0, stream>>>(dstate, w_out, b_out, (float*)d_out);
}
